// Net_14370960573087
// MI455X (gfx1250) — compile-verified, matrix operand order checked by reading, not run on hardware
//
#include <hip/hip_runtime.h>
#include <math.h>

// ---------------------------------------------------------------------------
// Types for CDNA5 WMMA
// ---------------------------------------------------------------------------
typedef __bf16 bf16_t;
typedef bf16_t v16bf __attribute__((ext_vector_type(16)));
typedef float  v8f   __attribute__((ext_vector_type(8)));

__device__ __forceinline__ bf16_t f2bf(float f) {
    unsigned u = __builtin_bit_cast(unsigned, f);
    unsigned r = u + 0x7FFFu + ((u >> 16) & 1u);   // round-to-nearest-even
    unsigned short h = (unsigned short)(r >> 16);
    return __builtin_bit_cast(bf16_t, h);
}

// ---------------------------------------------------------------------------
// Problem constants (B=2, N=8192 per reference)
// ---------------------------------------------------------------------------
#define PB   2
#define PN   8192
#define PM1  820          // ceil(0.1*8192)
#define PM2  41           // ceil(0.05*820)
#define KNB  64           // MAX_NEIGHBORS
#define E1   (PB*PM1*KNB) // 104960
#define E2   (PB*PM2*KNB) // 5248
#define BNEPS 1e-5f

static inline unsigned cdiv(long a, long b) { return (unsigned)((a + b - 1) / b); }

// ---------------------------------------------------------------------------
// WMMA GEMM:  C[M,N] = act(A[M,K] @ W[K,N] + bias)   (f32 in, bf16 MACs, f32 out)
// Block tile 64x128, BK=32, 128 threads = 4 waves, wave tile 32x64 (2x4 wmma).
// Fast path (full tile, K%4==0): vectorized b128 loads + packed b64 LDS stores,
// no per-element predication (EXEC stays full).
// ---------------------------------------------------------------------------
#define BM  64
#define BNT 128
#define BK  32

__global__ __launch_bounds__(128)
void k_gemm(const float* __restrict__ A, const float* __restrict__ W,
            const float* __restrict__ bias, float* __restrict__ C,
            int M, int K, int N, int relu) {
    __shared__ bf16_t As[BM][BK];      // 4 KB
    __shared__ bf16_t Bs[BNT][BK];     // 8 KB, transposed: [n][k]

    int t    = threadIdx.x;
    int wave = t >> 5;
    int lane = t & 31;
    int m0 = blockIdx.x * BM;
    int n0 = blockIdx.y * BNT;
    int wm = (wave >> 1) * 32;
    int wn = (wave & 1) * 64;

    v8f zero = 0.0f;
    v8f acc[2][4];
    #pragma unroll
    for (int i = 0; i < 2; ++i)
        #pragma unroll
        for (int j = 0; j < 4; ++j) acc[i][j] = zero;

    bool fullM = (m0 + BM <= M);
    bool k4    = ((K & 3) == 0);
    int ktiles = (K + BK - 1) / BK;

    for (int kb = 0; kb < ktiles; ++kb) {
        int k0 = kb * BK;
        if (fullM && k4 && (k0 + BK <= K)) {
            // ---- fast path: uniform, vectorized ----
            #pragma unroll
            for (int i = 0; i < 4; ++i) {          // A: 64x32 = 512 float4 chunks
                int id = t + i * 128;
                int r  = id >> 3;
                int kc = (id & 7) << 2;
                const float4 v = *(const float4*)(A + (size_t)(m0 + r) * K + (k0 + kc));
                union { bf16_t h[4]; unsigned long long u; } p;
                p.h[0] = f2bf(v.x); p.h[1] = f2bf(v.y);
                p.h[2] = f2bf(v.z); p.h[3] = f2bf(v.w);
                *(unsigned long long*)(&As[r][kc]) = p.u;
            }
            #pragma unroll
            for (int i = 0; i < 8; ++i) {          // B: 32x128 = 1024 4-k chunks
                int id = t + i * 128;
                int n  = id & 127;                 // n fastest -> coalesced
                int kc = (id >> 7) << 2;
                union { bf16_t h[4]; unsigned long long u; } p;
                #pragma unroll
                for (int j = 0; j < 4; ++j)
                    p.h[j] = f2bf(W[(size_t)(k0 + kc + j) * N + (n0 + n)]);
                *(unsigned long long*)(&Bs[n][kc]) = p.u;
            }
            if (kb + 1 < ktiles) {                 // prefetch next K tile
                __builtin_prefetch(A + (size_t)(m0 + (t >> 1)) * K + (k0 + BK), 0, 1);
                __builtin_prefetch(W + (size_t)(k0 + BK + (t >> 2)) * N + n0, 0, 1);
            }
        } else {
            // ---- slow path: per-element guards (small/odd-K layers only) ----
            for (int i = t; i < BM * BK; i += 128) {
                int r = i >> 5, c = i & 31;
                int gr = m0 + r, gc = k0 + c;
                float v = (gr < M && gc < K) ? A[(size_t)gr * K + gc] : 0.0f;
                As[r][c] = f2bf(v);
            }
            for (int i = t; i < BK * BNT; i += 128) {
                int r = i >> 7, c = i & 127;       // r = k, c = n
                int gk = k0 + r, gn = n0 + c;
                float v = (gk < K && gn < N) ? W[(size_t)gk * N + gn] : 0.0f;
                Bs[c][r] = f2bf(v);
            }
        }
        __syncthreads();

        int half = (lane >> 4) << 3;   // 0 for lanes 0-15, 8 for lanes 16-31
        int lr   = lane & 15;
        v16bf af[2], bfr[4];
        #pragma unroll
        for (int ti = 0; ti < 2; ++ti) {
            int row = wm + 16 * ti + lr;
            #pragma unroll
            for (int j = 0; j < 8; ++j) {
                af[ti][j]     = As[row][half + j];
                af[ti][8 + j] = As[row][16 + half + j];
            }
        }
        #pragma unroll
        for (int tj = 0; tj < 4; ++tj) {
            int col = wn + 16 * tj + lr;
            #pragma unroll
            for (int j = 0; j < 8; ++j) {
                bfr[tj][j]     = Bs[col][half + j];
                bfr[tj][8 + j] = Bs[col][16 + half + j];
            }
        }
        #pragma unroll
        for (int ti = 0; ti < 2; ++ti)
            #pragma unroll
            for (int tj = 0; tj < 4; ++tj)
                acc[ti][tj] = __builtin_amdgcn_wmma_f32_16x16x32_bf16(
                    false, af[ti], false, bfr[tj], (short)0, acc[ti][tj], false, false);
        __syncthreads();
    }

    int rbase = (lane >> 4) << 3;   // D layout: lanes 16-31 hold rows +8
    int nlane = lane & 15;
    #pragma unroll
    for (int ti = 0; ti < 2; ++ti)
        #pragma unroll
        for (int tj = 0; tj < 4; ++tj) {
            int col = n0 + wn + 16 * tj + nlane;
            #pragma unroll
            for (int r = 0; r < 8; ++r) {
                int row = m0 + wm + 16 * ti + rbase + r;
                if (row < M && col < N) {
                    float v = acc[ti][tj][r];
                    if (bias) v += bias[col];
                    if (relu) v = fmaxf(v, 0.0f);
                    C[(size_t)row * N + col] = v;
                }
            }
        }
}

// ---------------------------------------------------------------------------
// Farthest point sampling: one block per batch; deterministic start at 0.
// ---------------------------------------------------------------------------
__global__ __launch_bounds__(1024)
void k_fps(const float* __restrict__ pos, int n, int m, int* __restrict__ outIdx) {
    int b = blockIdx.x;
    pos    += (size_t)b * n * 3;
    outIdx += (size_t)b * m;
    __shared__ float mind[PN];
    __shared__ float rv[1024];
    __shared__ int   ri[1024];
    __shared__ int   s_last;
    int t = threadIdx.x;
    for (int i = t; i < n; i += 1024) mind[i] = 3.4e38f;
    if (t == 0) { s_last = 0; outIdx[0] = 0; }
    __syncthreads();
    for (int it = 1; it < m; ++it) {
        int last = s_last;
        float lx = pos[last*3+0], ly = pos[last*3+1], lz = pos[last*3+2];
        float bv = -1.0f; int bi = 0x7fffffff;
        for (int i = t; i < n; i += 1024) {
            float dx = pos[i*3+0]-lx, dy = pos[i*3+1]-ly, dz = pos[i*3+2]-lz;
            float d  = dx*dx + dy*dy + dz*dz;
            float mn = fminf(mind[i], d);
            mind[i] = mn;
            if (mn > bv) { bv = mn; bi = i; }   // strict > keeps lowest index
        }
        rv[t] = bv; ri[t] = bi;
        __syncthreads();
        for (int s = 512; s > 0; s >>= 1) {
            if (t < s) {
                if (rv[t+s] > rv[t] || (rv[t+s] == rv[t] && ri[t+s] < ri[t])) {
                    rv[t] = rv[t+s]; ri[t] = ri[t+s];
                }
            }
            __syncthreads();
        }
        if (t == 0) { s_last = ri[0]; outIdx[it] = ri[0]; }
        __syncthreads();
    }
}

// gather sampled positions: out[q] = pos[b, idx[q]]
__global__ void k_gatherpos(const float* __restrict__ pos, const int* __restrict__ idx,
                            int n, int m, float* __restrict__ out) {
    int i = blockIdx.x * blockDim.x + threadIdx.x;
    if (i >= PB * m) return;
    int b = i / m;
    int j = idx[i];
    out[i*3+0] = pos[((size_t)b*n + j)*3 + 0];
    out[i*3+1] = pos[((size_t)b*n + j)*3 + 1];
    out[i*3+2] = pos[((size_t)b*n + j)*3 + 2];
}

// ---------------------------------------------------------------------------
// Radius / K-nearest within r: one block per query, 64 min-extraction rounds.
// ---------------------------------------------------------------------------
__global__ __launch_bounds__(256)
void k_radius(const float* __restrict__ pos, int n, const float* __restrict__ qpos,
              int mq, float r2, int* __restrict__ nidx, unsigned char* __restrict__ mask) {
    int q = blockIdx.x;            // over PB*mq
    int b = q / mq;
    const float* P = pos + (size_t)b * n * 3;
    float qx = qpos[q*3+0], qy = qpos[q*3+1], qz = qpos[q*3+2];
    __shared__ float d2[PN];
    __shared__ float rv[256];
    __shared__ int   ri[256];
    int t = threadIdx.x;
    for (int i = t; i < n; i += 256) {
        float dx = P[i*3+0]-qx, dy = P[i*3+1]-qy, dz = P[i*3+2]-qz;
        d2[i] = dx*dx + dy*dy + dz*dz;
    }
    __syncthreads();
    for (int j = 0; j < KNB; ++j) {
        float bv = 3.0e38f; int bi = 0x7fffffff;
        for (int i = t; i < n; i += 256) {
            float v = d2[i];
            if (v < bv) { bv = v; bi = i; }
        }
        rv[t] = bv; ri[t] = bi;
        __syncthreads();
        for (int s = 128; s > 0; s >>= 1) {
            if (t < s) {
                if (rv[t+s] < rv[t] || (rv[t+s] == rv[t] && ri[t+s] < ri[t])) {
                    rv[t] = rv[t+s]; ri[t] = ri[t+s];
                }
            }
            __syncthreads();
        }
        int sel = ri[0]; float val = rv[0];
        bool ok = (val <= r2) && (sel < n);
        if (t == 0) {
            nidx[(size_t)q*KNB + j] = ok ? sel : 0;
            mask[(size_t)q*KNB + j] = ok ? 1 : 0;
            if (ok) d2[sel] = 3.4e38f;
        }
        __syncthreads();
    }
}

// SA1 message: rel positions only, cols=3
__global__ void k_msg1(const float* __restrict__ pos, const float* __restrict__ pos1,
                       const int* __restrict__ nidx, float* __restrict__ out) {
    int e = blockIdx.x * blockDim.x + threadIdx.x;
    if (e >= E1) return;
    int q = e / KNB;
    int b = q / PM1;
    int j = nidx[e];
    out[(size_t)e*3+0] = pos[((size_t)b*PN + j)*3 + 0] - pos1[q*3+0];
    out[(size_t)e*3+1] = pos[((size_t)b*PN + j)*3 + 1] - pos1[q*3+1];
    out[(size_t)e*3+2] = pos[((size_t)b*PN + j)*3 + 2] - pos1[q*3+2];
}

// SA2 message: [x1(512) | rel(3)], cols=515
__global__ void k_msg2(const float* __restrict__ pos1, const float* __restrict__ pos2,
                       const float* __restrict__ x1, const int* __restrict__ nidx,
                       float* __restrict__ out) {
    const int COLS = 515;
    long idx = (long)blockIdx.x * blockDim.x + threadIdx.x;
    if (idx >= (long)E2 * COLS) return;
    int e = (int)(idx / COLS), c = (int)(idx % COLS);
    int q = e / KNB;
    int b = q / PM2;
    int j = nidx[e];
    if (c < 512)
        out[idx] = x1[((size_t)(b*PM1 + j))*512 + c];
    else
        out[idx] = pos1[(size_t)(b*PM1 + j)*3 + (c-512)] - pos2[(size_t)q*3 + (c-512)];
}

// ---------------------------------------------------------------------------
// BatchNorm (training stats), two-stage coalesced reduction, opt. mask weight.
// ---------------------------------------------------------------------------
#define BNCHUNK 1024

__global__ __launch_bounds__(256)
void k_bnstats_part(const float* __restrict__ h, const unsigned char* __restrict__ mask,
                    int rows, int C, float* __restrict__ pS, float* __restrict__ pS2,
                    float* __restrict__ pCnt) {
    int chunk = blockIdx.x;
    int r0 = chunk * BNCHUNK;
    int r1 = min(r0 + BNCHUNK, rows);
    int t  = threadIdx.x;
    // valid-row count for this chunk
    float cnt = 0.f;
    if (mask) {
        for (int r = r0 + t; r < r1; r += 256) cnt += (float)mask[r];
    }
    __shared__ float sc[256];
    sc[t] = cnt;
    __syncthreads();
    for (int s = 128; s > 0; s >>= 1) {
        if (t < s) sc[t] += sc[t+s];
        __syncthreads();
    }
    if (t == 0) pCnt[chunk] = mask ? sc[0] : (float)(r1 - r0);
    // per-channel partials (lane-consecutive channels -> coalesced)
    for (int c = t; c < C; c += 256) {
        float s = 0.f, s2 = 0.f;
        for (int r = r0; r < r1; ++r) {
            float w = mask ? (float)mask[r] : 1.0f;
            float v = h[(size_t)r * C + c];
            s  += w * v;
            s2 += w * v * v;
        }
        pS [(size_t)chunk * C + c] = s;
        pS2[(size_t)chunk * C + c] = s2;
    }
}

__global__ void k_bnstats_final(const float* __restrict__ pS, const float* __restrict__ pS2,
                                const float* __restrict__ pCnt, int nch, int C,
                                float* __restrict__ mean, float* __restrict__ var) {
    int c = blockIdx.x * blockDim.x + threadIdx.x;
    if (c >= C) return;
    float s = 0.f, s2 = 0.f, cw = 0.f;
    for (int k = 0; k < nch; ++k) {
        s  += pS [(size_t)k * C + c];
        s2 += pS2[(size_t)k * C + c];
        cw += pCnt[k];
    }
    float denom = fmaxf(cw, 1.0f);
    float mu = s / denom;
    mean[c] = mu;
    var[c]  = fmaxf(s2 / denom - mu * mu, 0.0f);
}

__global__ void k_bnapply(float* __restrict__ h, long total, int C,
                          const float* __restrict__ mean, const float* __restrict__ var,
                          const float* __restrict__ g, const float* __restrict__ beta, int relu) {
    long i = (long)blockIdx.x * blockDim.x + threadIdx.x;
    if (i >= total) return;
    int c = (int)(i % C);
    float v = (h[i] - mean[c]) * rsqrtf(var[c] + BNEPS) * g[c] + beta[c];
    h[i] = relu ? fmaxf(v, 0.0f) : v;
}

// masked max over K neighbors (invalid -> -inf, like reference)
__global__ void k_maxpool(const float* __restrict__ h, const unsigned char* __restrict__ mask,
                          int C, float* __restrict__ out) {
    int q = blockIdx.x;
    for (int c = threadIdx.x; c < C; c += blockDim.x) {
        float v = -INFINITY;
        for (int k = 0; k < KNB; ++k)
            if (mask[(size_t)q*KNB + k])
                v = fmaxf(v, h[((size_t)q*KNB + k)*C + c]);
        out[(size_t)q*C + c] = v;
    }
}

// global max pool over points per batch
__global__ void k_gmax(const float* __restrict__ h, int m, int C, float* __restrict__ out) {
    int b = blockIdx.x;
    for (int c = threadIdx.x; c < C; c += blockDim.x) {
        float v = -INFINITY;
        for (int i = 0; i < m; ++i)
            v = fmaxf(v, h[((size_t)b*m + i)*C + c]);
        out[(size_t)b*C + c] = v;
    }
}

// SA3 input: [x2(1024) | pos2(3)]
__global__ void k_cat_sa3(const float* __restrict__ x2, const float* __restrict__ pos2,
                          float* __restrict__ out) {
    const int COLS = 1027;
    long idx = (long)blockIdx.x * blockDim.x + threadIdx.x;
    if (idx >= (long)PB*PM2*COLS) return;
    int r = (int)(idx / COLS), c = (int)(idx % COLS);
    out[idx] = (c < 1024) ? x2[(size_t)r*1024 + c] : pos2[(size_t)r*3 + (c-1024)];
}

// FP3 input: [x3 broadcast (2048) | x2 (1024)]
__global__ void k_cat_fp3(const float* __restrict__ x3, const float* __restrict__ x2,
                          float* __restrict__ out) {
    const int COLS = 3072;
    long idx = (long)blockIdx.x * blockDim.x + threadIdx.x;
    if (idx >= (long)PB*PM2*COLS) return;
    int r = (int)(idx / COLS), c = (int)(idx % COLS);
    int b = r / PM2;
    out[idx] = (c < 2048) ? x3[(size_t)b*2048 + c] : x2[(size_t)r*1024 + (c-2048)];
}

// copy skip features into concat buffer columns
__global__ void k_copy_cols(float* __restrict__ dst, int dstLd, int dstOff,
                            const float* __restrict__ src, int srcC, int rows) {
    long idx = (long)blockIdx.x * blockDim.x + threadIdx.x;
    if (idx >= (long)rows * srcC) return;
    int r = (int)(idx / srcC), c = (int)(idx % srcC);
    dst[(size_t)r*dstLd + dstOff + c] = src[idx];
}

// 3-NN select + inverse-squared-distance weights
__global__ void k_knn3(const float* __restrict__ spos, int M,
                       const float* __restrict__ dpos, int Nd,
                       int* __restrict__ kidx, float* __restrict__ kw) {
    int i = blockIdx.x * blockDim.x + threadIdx.x;
    if (i >= PB * Nd) return;
    int b = i / Nd;
    float px = dpos[i*3+0], py = dpos[i*3+1], pz = dpos[i*3+2];
    float d0 = 3.4e38f, d1 = 3.4e38f, d2v = 3.4e38f;
    int i0 = 0, i1 = 0, i2 = 0;
    for (int j = 0; j < M; ++j) {
        const float* s = spos + ((size_t)b*M + j)*3;
        float dx = s[0]-px, dy = s[1]-py, dz = s[2]-pz;
        float d = dx*dx + dy*dy + dz*dz;
        if (d < d0)       { d2v = d1; i2 = i1; d1 = d0; i1 = i0; d0 = d; i0 = j; }
        else if (d < d1)  { d2v = d1; i2 = i1; d1 = d;  i1 = j; }
        else if (d < d2v) { d2v = d;  i2 = j; }
    }
    kidx[(size_t)i*3+0] = i0; kw[(size_t)i*3+0] = 1.0f / fmaxf(d0,  1e-16f);
    kidx[(size_t)i*3+1] = i1; kw[(size_t)i*3+1] = 1.0f / fmaxf(d1,  1e-16f);
    kidx[(size_t)i*3+2] = i2; kw[(size_t)i*3+2] = 1.0f / fmaxf(d2v, 1e-16f);
}

__global__ void k_knn_apply(const float* __restrict__ x, int M, int C,
                            const int* __restrict__ kidx, const float* __restrict__ kw,
                            float* __restrict__ dst, int dstLd, int dstOff, int Nd) {
    int i = blockIdx.x;     // over PB*Nd
    int b = i / Nd;
    int j0 = kidx[(size_t)i*3+0], j1 = kidx[(size_t)i*3+1], j2 = kidx[(size_t)i*3+2];
    float w0 = kw[(size_t)i*3+0], w1 = kw[(size_t)i*3+1], w2 = kw[(size_t)i*3+2];
    float inv = 1.0f / (w0 + w1 + w2);
    const float* x0  = x + ((size_t)b*M + j0)*C;
    const float* x1p = x + ((size_t)b*M + j1)*C;
    const float* x2p = x + ((size_t)b*M + j2)*C;
    for (int c = threadIdx.x; c < C; c += blockDim.x)
        dst[(size_t)i*dstLd + dstOff + c] = (w0*x0[c] + w1*x1p[c] + w2*x2p[c]) * inv;
}

// head: logits = h@W2 + b2; log_softmax; store transposed [1,4,rows]
__global__ void k_head(const float* __restrict__ h, const float* __restrict__ W,
                       const float* __restrict__ bias, int rows, float* __restrict__ out) {
    int r = blockIdx.x * blockDim.x + threadIdx.x;
    if (r >= rows) return;
    float acc[4] = { bias[0], bias[1], bias[2], bias[3] };
    const float* hr = h + (size_t)r * 1024;
    for (int k = 0; k < 1024; ++k) {
        float v = hr[k];
        acc[0] += v * W[k*4+0];
        acc[1] += v * W[k*4+1];
        acc[2] += v * W[k*4+2];
        acc[3] += v * W[k*4+3];
    }
    float mx = fmaxf(fmaxf(acc[0], acc[1]), fmaxf(acc[2], acc[3]));
    float se = 0.f;
    for (int c = 0; c < 4; ++c) se += expf(acc[c] - mx);
    float lse = mx + logf(se);
    for (int c = 0; c < 4; ++c)
        out[(size_t)c * rows + r] = acc[c] - lse;
}

// ---------------------------------------------------------------------------
// Host orchestration
// ---------------------------------------------------------------------------
extern "C" void kernel_launch(void* const* d_in, const int* in_sizes, int n_in,
                              void* d_out, int out_size, void* d_ws, size_t ws_size,
                              hipStream_t stream) {
    const float* pos = (const float*)d_in[0];
    auto P = [&](int i) -> const float* { return (const float*)d_in[i]; };

    // workspace bump allocator
    char* wsp = (char*)d_ws;
    size_t off = 0;
    auto alloc = [&](size_t bytes) -> void* {
        void* p = wsp + off;
        off = (off + bytes + 255) & ~(size_t)255;
        return p;
    };
    float* bufA  = (float*)alloc((size_t)E1 * 512 * 4);
    float* bufB  = (float*)alloc((size_t)E1 * 512 * 4);
    float* bufC  = (float*)alloc((size_t)E2 * 515 * 4);
    int*   idx1  = (int*)  alloc((size_t)PB*PM1*4);
    float* pos1  = (float*)alloc((size_t)PB*PM1*3*4);
    int*   nidx1 = (int*)  alloc((size_t)E1*4);
    unsigned char* mask1 = (unsigned char*)alloc(E1);
    float* x1    = (float*)alloc((size_t)PB*PM1*512*4);
    int*   idx2  = (int*)  alloc((size_t)PB*PM2*4);
    float* pos2  = (float*)alloc((size_t)PB*PM2*3*4);
    int*   nidx2 = (int*)  alloc((size_t)E2*4);
    unsigned char* mask2 = (unsigned char*)alloc(E2);
    float* x2    = (float*)alloc((size_t)PB*PM2*1024*4);
    float* x3    = (float*)alloc((size_t)PB*2048*4);
    float* meanB = (float*)alloc(2048*4);
    float* varB  = (float*)alloc(2048*4);
    int*   kidx  = (int*)  alloc((size_t)PB*PN*3*4);
    float* kw    = (float*)alloc((size_t)PB*PN*3*4);
    const int NCHMAX = (E1 + BNCHUNK - 1) / BNCHUNK;   // 103
    float* partS  = (float*)alloc((size_t)NCHMAX * 2048 * 4);
    float* partS2 = (float*)alloc((size_t)NCHMAX * 2048 * 4);
    float* partC  = (float*)alloc((size_t)NCHMAX * 4);

    auto GEMM = [&](const float* A, const float* Wp, const float* bp, float* Cp,
                    int M, int K, int N, int relu) {
        dim3 g(cdiv(M, BM), (unsigned)(N / BNT));
        k_gemm<<<g, 128, 0, stream>>>(A, Wp, bp, Cp, M, K, N, relu);
    };
    auto BN = [&](float* h, const unsigned char* mask, int rows, int C,
                  const float* g, const float* beta, int relu) {
        int nch = (int)cdiv(rows, BNCHUNK);
        k_bnstats_part<<<nch, 256, 0, stream>>>(h, mask, rows, C, partS, partS2, partC);
        k_bnstats_final<<<cdiv(C, 256), 256, 0, stream>>>(partS, partS2, partC, nch, C, meanB, varB);
        long tot = (long)rows * C;
        k_bnapply<<<cdiv(tot, 256), 256, 0, stream>>>(h, tot, C, meanB, varB, g, beta, relu);
    };

    // ---------------- SA1 ----------------
    k_fps<<<PB, 1024, 0, stream>>>(pos, PN, PM1, idx1);
    k_gatherpos<<<cdiv(PB*PM1, 256), 256, 0, stream>>>(pos, idx1, PN, PM1, pos1);
    k_radius<<<PB*PM1, 256, 0, stream>>>(pos, PN, pos1, PM1, 0.04f, nidx1, mask1);
    k_msg1<<<cdiv(E1, 256), 256, 0, stream>>>(pos, pos1, nidx1, bufA);
    GEMM(bufA, P(3),  P(4),  bufB, E1, 3,   128, 1); BN(bufB, mask1, E1, 128, P(5),  P(6),  0);
    GEMM(bufB, P(7),  P(8),  bufA, E1, 128, 256, 1); BN(bufA, mask1, E1, 256, P(9),  P(10), 0);
    GEMM(bufA, P(11), P(12), bufB, E1, 256, 512, 1); BN(bufB, mask1, E1, 512, P(13), P(14), 0);
    k_maxpool<<<PB*PM1, 256, 0, stream>>>(bufB, mask1, 512, x1);

    // ---------------- SA2 ----------------
    k_fps<<<PB, 1024, 0, stream>>>(pos1, PM1, PM2, idx2);
    k_gatherpos<<<cdiv(PB*PM2, 256), 256, 0, stream>>>(pos1, idx2, PM1, PM2, pos2);
    k_radius<<<PB*PM2, 256, 0, stream>>>(pos1, PM1, pos2, PM2, 0.16f, nidx2, mask2);
    k_msg2<<<cdiv((long)E2*515, 256), 256, 0, stream>>>(pos1, pos2, x1, nidx2, bufA);
    GEMM(bufA, P(15), P(16), bufB, E2, 515,  512,  1); BN(bufB, mask2, E2, 512,  P(17), P(18), 0);
    GEMM(bufB, P(19), P(20), bufA, E2, 512,  1024, 1); BN(bufA, mask2, E2, 1024, P(21), P(22), 0);
    GEMM(bufA, P(23), P(24), bufB, E2, 1024, 1024, 1); BN(bufB, mask2, E2, 1024, P(25), P(26), 0);
    k_maxpool<<<PB*PM2, 256, 0, stream>>>(bufB, mask2, 1024, x2);

    // ---------------- SA3 (global) ----------------
    const int R3 = PB * PM2;   // 82
    k_cat_sa3<<<cdiv((long)R3*1027, 256), 256, 0, stream>>>(x2, pos2, bufC);
    GEMM(bufC, P(27), P(28), bufA, R3, 1027, 1024, 1); BN(bufA, nullptr, R3, 1024, P(29), P(30), 0);
    GEMM(bufA, P(31), P(32), bufB, R3, 1024, 2048, 1); BN(bufB, nullptr, R3, 2048, P(33), P(34), 0);
    GEMM(bufB, P(35), P(36), bufA, R3, 2048, 2048, 1); BN(bufA, nullptr, R3, 2048, P(37), P(38), 0);
    k_gmax<<<PB, 256, 0, stream>>>(bufA, PM2, 2048, x3);

    // ---------------- FP3 (k=1 broadcast) ----------------
    k_cat_fp3<<<cdiv((long)R3*3072, 256), 256, 0, stream>>>(x3, x2, bufC);
    GEMM(bufC, P(39), P(40), bufA, R3, 3072, 1024, 1); BN(bufA, nullptr, R3, 1024, P(41), P(42), 0);
    GEMM(bufA, P(43), P(44), bufB, R3, 1024, 1024, 1); BN(bufB, nullptr, R3, 1024, P(45), P(46), 0);

    // ---------------- FP2 (k=3) ----------------
    const int R2 = PB * PM1;   // 1640
    k_knn3<<<cdiv(R2, 256), 256, 0, stream>>>(pos2, PM2, pos1, PM1, kidx, kw);
    k_knn_apply<<<R2, 256, 0, stream>>>(bufB, PM2, 1024, kidx, kw, bufC, 1536, 0, PM1);
    k_copy_cols<<<cdiv((long)R2*512, 256), 256, 0, stream>>>(bufC, 1536, 1024, x1, 512, R2);
    GEMM(bufC, P(47), P(48), bufA, R2, 1536, 1024, 1); BN(bufA, nullptr, R2, 1024, P(49), P(50), 0);
    GEMM(bufA, P(51), P(52), bufB, R2, 1024, 1024, 1); BN(bufB, nullptr, R2, 1024, P(53), P(54), 0);

    // ---------------- FP1 (k=3) ----------------
    const int R1 = PB * PN;    // 16384
    k_knn3<<<cdiv(R1, 256), 256, 0, stream>>>(pos1, PM1, pos, PN, kidx, kw);
    k_knn_apply<<<R1, 256, 0, stream>>>(bufB, PM1, 1024, kidx, kw, bufA, 1024, 0, PN);
    GEMM(bufA, P(55), P(56), bufB, R1, 1024, 1024, 1); BN(bufB, nullptr, R1, 1024, P(57), P(58), 0);
    GEMM(bufB, P(59), P(60), bufA, R1, 1024, 1024, 1); BN(bufA, nullptr, R1, 1024, P(61), P(62), 0);

    // ---------------- Head ----------------
    GEMM(bufA, P(63), P(64), bufB, R1, 1024, 1024, 0); BN(bufB, nullptr, R1, 1024, P(65), P(66), 1);
    k_head<<<cdiv(R1, 256), 256, 0, stream>>>(bufB, P(67), P(68), R1, (float*)d_out);
}
